// CrissCrossAttention_23536420782567
// MI455X (gfx1250) — matrix operand order repaired and checked by reading, not compiled, not run on hardware
//
#include <hip/hip_runtime.h>
#include <stdint.h>

// ---------------------------------------------------------------------------
// Criss-cross attention for MI455X (gfx1250), bf16 WMMA + fp32 accumulate.
// Stage 1: QKV 1x1-conv GEMMs  (fp32 in -> bf16 out to workspace)
// Stage 2: column (H) attention -> out_h fp32 workspace  (TDM-gathered V tiles)
// Stage 3: row (W) attention + fused final compose -> d_out
// ---------------------------------------------------------------------------

typedef __attribute__((ext_vector_type(16))) __bf16 v16bf;
typedef __attribute__((ext_vector_type(8)))  __bf16 v8bf;
typedef __attribute__((ext_vector_type(8)))  float  v8f;
typedef __attribute__((ext_vector_type(4)))  unsigned int v4u;
typedef __attribute__((ext_vector_type(8)))  int v8i;
typedef __attribute__((ext_vector_type(4)))  int v4i;

constexpr int kB = 16, kC = 512, kInter = 64, kH = 64, kW = 64;
constexpr int kHW = kH * kW; // 4096

#define HAVE_TDM (__has_builtin(__builtin_amdgcn_tensor_load_to_lds) && \
                  __has_builtin(__builtin_amdgcn_s_wait_tensorcnt))

// --- WMMA fragment helpers (CDNA5 wave32 layouts, CDNA5 ISA) -----
// A-fragment (16x32 bf16, MxK) from a K-contiguous row-major tile: per lane the
// 16 elements are two contiguous 8-element runs -> two 16-byte loads.
static __device__ __forceinline__ v16bf frag_a_k1(const __bf16* base, int mStride) {
  const int lane = threadIdx.x & 31;
  const __bf16* p = base + (lane & 15) * mStride + ((lane & 16) ? 8 : 0);
  v8bf lo = *(const v8bf*)p;
  v8bf hi = *(const v8bf*)(p + 16);
  v16bf a;
#pragma unroll
  for (int r = 0; r < 8; ++r) { a[r] = lo[r]; a[8 + r] = hi[r]; }
  return a;
}

// B-fragment (32x16 bf16, KxN) from a fragment-ordered tile: element (k,n) was
// staged at [lane = n | ((k&8)<<1)][elem = (k&7) | ((k&16)>>1)], so each lane reads 32 contiguous B.
static __device__ __forceinline__ v16bf frag_b_packed(const __bf16* tile) {
  const int lane = threadIdx.x & 31;
  return *(const v16bf*)(tile + lane * 16);
}

static __device__ __forceinline__ v8f wmma_bf16(v16bf a, v16bf b, v8f c) {
  return __builtin_amdgcn_wmma_f32_16x16x32_bf16(false, a, false, b, (short)0, c,
                                                 false, false);
}

#if HAVE_TDM
// TDM descriptor: gather a [nZ rows x nY elems] 16-bit tile where successive Y
// elements are strideY apart and successive Z rows are strideZ apart (element
// units), packed K-contiguous into LDS. tile = [x=1][y=nY][z=nZ], data_size=2B.
static __device__ __forceinline__ void
tdm_gather_2d_b16(uint32_t lds_addr, const void* gptr,
                  uint32_t nY, uint32_t nZ, uint32_t strideY, uint32_t strideZ) {
  uint64_t ga = (uint64_t)(uintptr_t)gptr;
  v4u g0;
  g0[0] = 1u;                                                // count=1, user D#
  g0[1] = lds_addr;                                          // lds byte address
  g0[2] = (uint32_t)ga;                                      // global addr lo
  g0[3] = (uint32_t)((ga >> 32) & 0x01FFFFFFu) | (2u << 30); // addr[56:32]|type=2
  v8i g1;
  g1[0] = (int)(1u << 16);                                   // data_size = 2 bytes
  g1[1] = (int)(1u << 16);                                   // tensor_dim0 = 1
  g1[2] = (int)((nY & 0xFFFFu) << 16);                       // tensor_dim1 lo16
  g1[3] = (int)(1u << 16);                                   // tile_dim0 = 1
  g1[4] = (int)((nY & 0xFFFFu) | ((nZ & 0xFFFFu) << 16));    // tile_dim1|tile_dim2
  g1[5] = (int)strideY;                                      // dim0_stride lo32
  g1[6] = (int)((strideZ & 0xFFFFu) << 16);                  // dim1_stride lo16
  g1[7] = (int)(strideZ >> 16);                              // dim1_stride [47:16]
  v4i g2; g2[0] = (int)nZ; g2[1] = 0; g2[2] = 0; g2[3] = 0;  // tensor_dim2
  v4i g3 = {0, 0, 0, 0};
  v8i g4 = {0, 0, 0, 0, 0, 0, 0, 0};                         // unused extra group
  __builtin_amdgcn_tensor_load_to_lds(g0, g1, g2, g3, g4, 0);
}

static __device__ __forceinline__ void tdm_wait(int mi) {
  // s_wait_tensorcnt needs a frontend-constant immediate; mi is wave-uniform
  // and constant after unrolling, so this switch folds to a single wait.
  switch (mi) {
    case 0: __builtin_amdgcn_s_wait_tensorcnt(3); break;
    case 1: __builtin_amdgcn_s_wait_tensorcnt(2); break;
    case 2: __builtin_amdgcn_s_wait_tensorcnt(1); break;
    default: __builtin_amdgcn_s_wait_tensorcnt(0); break;
  }
}
#endif

// ---------------------------------------------------------------------------
// Stage 1: OUT[b, m, p] = sum_c Wm[m, c] * x[b, c, p] + bias[m], stored bf16.
// Block = 256 threads = 8 waves (2M x 4N), tile 32M x 64N, K staged 32-wide.
// ---------------------------------------------------------------------------
__global__ void __launch_bounds__(256)
qkv_gemm_kernel(const float* __restrict__ x, const float* __restrict__ Wm,
                const float* __restrict__ bias, __bf16* __restrict__ out, int Mrows) {
  constexpr int LDW = 40; // multiple of 8 -> 16B-aligned fragment runs
  __shared__ alignas(32) __bf16 sW[32 * LDW];     // A: row-major 32m x 32k
  __shared__ alignas(32) __bf16 sXf[4 * 32 * 16]; // B: 4 frag-ordered 32k x 16n

  const int tid = threadIdx.x;
  const int waveId = tid >> 5, lane = tid & 31;
  const int waveM = waveId >> 2, waveN = waveId & 3;
  const int n0 = blockIdx.x * 64, m0 = blockIdx.y * 32, b = blockIdx.z;
  const float* xb = x + (size_t)b * kC * kHW;

  v8f acc = {};
  for (int k0 = 0; k0 < kC; k0 += 32) {
#pragma unroll
    for (int t = 0; t < 2; ++t) {                 // W tile: 512 k-pairs
      int pi = tid + t * 256;
      int m = pi >> 4, k = (pi & 15) * 2;
      const float* src = &Wm[(size_t)(m0 + m) * kC + k0 + k];
      sW[m * LDW + k]     = (__bf16)src[0];
      sW[m * LDW + k + 1] = (__bf16)src[1];
    }
#pragma unroll
    for (int t = 0; t < 4; ++t) {                 // X tile: 1024 k-pairs, frag order
      int pi = tid + t * 256;
      int n = pi & 63, k = (pi >> 6) * 2;
      float f0 = xb[(size_t)(k0 + k) * kHW + n0 + n];
      float f1 = xb[(size_t)(k0 + k + 1) * kHW + n0 + n];
      __bf16* dst = &sXf[(((n >> 4) * 32) + ((n & 15) | ((k & 8) << 1))) * 16 + ((k & 7) | ((k & 16) >> 1))];
      dst[0] = (__bf16)f0;
      dst[1] = (__bf16)f1;
    }
    __syncthreads();
    v16bf a  = frag_a_k1(sW + waveM * 16 * LDW, LDW);
    v16bf bb = frag_b_packed(sXf + waveN * 32 * 16);
    acc = wmma_bf16(a, bb, acc);
    __syncthreads();
  }
  const int mBase = m0 + waveM * 16 + ((lane >> 4) << 3);
  const int nCol  = n0 + waveN * 16 + (lane & 15);
#pragma unroll
  for (int d = 0; d < 8; ++d) {
    int m = mBase + d;
    out[((size_t)b * Mrows + m) * kHW + nCol] = (__bf16)(acc[d] + bias[m]);
  }
}

// ---------------------------------------------------------------------------
// Stage 2/3: axis attention. DIR=0: over H at fixed w (writes out_h to ws),
// V tiles gathered by the Tensor Data Mover. DIR=1: over W at fixed h (fuses
// gamma*(out_h+out_v)+x into d_out), V rows contiguous -> direct b128 loads.
// One 256-thread block (8 waves) per (b, slice).
// ---------------------------------------------------------------------------
template <int DIR>
__global__ void __launch_bounds__(256)
axis_attn_kernel(const __bf16* __restrict__ qb, const __bf16* __restrict__ kbuf,
                 const __bf16* __restrict__ vb, float* __restrict__ outH,
                 const float* __restrict__ x, const float* __restrict__ gamma,
                 float* __restrict__ out) {
  constexpr int posStride   = (DIR == 0) ? kW : 1;
  constexpr int sliceStride = (DIR == 0) ? 1 : kW;
  constexpr int LDQ = 72, LDE = 68;

  __shared__ alignas(32) __bf16 sQt[64 * LDQ];        // q^T [i][c] (A); aliased below
  __shared__ alignas(32) __bf16 sKf[2 * 4 * 32 * 16]; // frag-ordered energy-B tiles
  __shared__ alignas(32) float  sE[64 * LDE];
#if HAVE_TDM
  __shared__ alignas(32) __bf16 sV[(DIR == 0) ? 8 * 4 * 16 * 64 : 1]; // per-wave V chunks
#endif
  __bf16* sAf = sQt; // frag-ordered attn^T B tiles (reuses sQt after barrier)

  const int tid = threadIdx.x;
  const int waveId = tid >> 5, lane = tid & 31;
  const int s = blockIdx.x, b = blockIdx.y;

  // ---- stage q (row-major transposed) and k (fragment order) ----
  const size_t qkBase = (size_t)b * kInter * kHW + (size_t)s * sliceStride;
#pragma unroll
  for (int t = 0; t < 8; ++t) {  // 2048 (c,c+1) pairs
    int pi = tid + t * 256;
    int pos = pi & 63, c = (pi >> 6) * 2;
    size_t g = qkBase + (size_t)c * kHW + (size_t)pos * posStride;
    sQt[pos * LDQ + c]     = qb[g];
    sQt[pos * LDQ + c + 1] = qb[g + kHW];
    int kk = c & 31;
    __bf16* dst = &sKf[((((c >> 5) * 4) + (pos >> 4)) * 32 +
                        ((pos & 15) | ((kk & 8) << 1))) * 16 + ((kk & 7) | ((kk & 16) >> 1))];
    dst[0] = kbuf[g];
    dst[1] = kbuf[g + kHW];
  }
  __syncthreads();

  // ---- energy: E[i][j] = sum_c qT[i][c] * k[c][j] (16 tiles, 2 per wave) ----
#pragma unroll
  for (int tt = 0; tt < 2; ++tt) {
    int t = waveId * 2 + tt;
    int ti = t >> 2, tj = t & 3;
    v8f e = {};
#pragma unroll
    for (int kc = 0; kc < 2; ++kc) {
      v16bf a  = frag_a_k1(sQt + ti * 16 * LDQ + kc * 32, LDQ);
      v16bf bb = frag_b_packed(sKf + ((kc * 4 + tj) * 32) * 16);
      e = wmma_bf16(a, bb, e);
    }
    int row = ti * 16 + ((lane >> 4) << 3);
    int col = tj * 16 + (lane & 15);
#pragma unroll
    for (int d = 0; d < 8; ++d) sE[(row + d) * LDE + col] = e[d];
  }
  __syncthreads();

  // ---- fp32 softmax over j; write attn^T in fragment order (aliases sQt) ----
#pragma unroll
  for (int rr = 0; rr < 8; ++rr) {
    int i = waveId + rr * 8;
    float e0 = sE[i * LDE + lane];
    float e1 = sE[i * LDE + lane + 32];
    float mx = fmaxf(e0, e1);
#pragma unroll
    for (int off = 16; off > 0; off >>= 1) mx = fmaxf(mx, __shfl_xor(mx, off, 32));
    e0 = __expf(e0 - mx);
    e1 = __expf(e1 - mx);
    float sum = e0 + e1;
#pragma unroll
    for (int off = 16; off > 0; off >>= 1) sum += __shfl_xor(sum, off, 32);
    float inv = 1.0f / sum;
    int nt = i >> 4, fl = (i & 15) | ((lane & 8) << 1), e = (lane & 7) | ((lane & 16) >> 1);
    sAf[((0 * 4 + nt) * 32 + fl) * 16 + e] = (__bf16)(e0 * inv); // j = lane
    sAf[((1 * 4 + nt) * 32 + fl) * 16 + e] = (__bf16)(e1 * inv); // j = lane+32
  }
  __syncthreads();

  // ---- AV: out[c][i] = sum_j v[c][j] * attn[i][j] ----
  const float g = (DIR == 1) ? gamma[0] : 0.0f;
  const size_t vSlice = (size_t)s * sliceStride;

#if HAVE_TDM
  if constexpr (DIR == 0) {
    // Each wave asynchronously gathers its 4 transposed [16c x 64j] V chunks.
    const unsigned waveU = (unsigned)__builtin_amdgcn_readfirstlane(waveId);
#pragma unroll
    for (int mi = 0; mi < 4; ++mi) {
      int ct = (int)waveU + mi * 8;
      const __bf16* gsrc = vb + ((size_t)b * kC + ct * 16) * kHW + vSlice;
      uint32_t ldsoff = (uint32_t)(uintptr_t)(const void*)&sV[(waveU * 4 + mi) * 1024];
      tdm_gather_2d_b16(ldsoff, gsrc, /*nY=*/64, /*nZ=*/16,
                        /*strideY=*/(uint32_t)posStride, /*strideZ=*/(uint32_t)kHW);
    }
  }
#endif

#pragma unroll
  for (int mi = 0; mi < 4; ++mi) {
    int c0 = (waveId + mi * 8) * 16;
    v16bf a0, a1;
    if constexpr (DIR == 1) {
      const __bf16* abase = vb + ((size_t)b * kC + c0) * kHW + vSlice;
      a0 = frag_a_k1(abase, kHW);
      a1 = frag_a_k1(abase + 32, kHW);
    } else {
#if HAVE_TDM
      tdm_wait(mi); // per-wave TENSORcnt is in-order: chunk mi is resident
      const __bf16* chunk = &sV[((unsigned)waveId * 4 + mi) * 1024];
      a0 = frag_a_k1(chunk, 64);
      a1 = frag_a_k1(chunk + 32, 64);
#else
      const __bf16* p = vb + ((size_t)b * kC + c0 + (lane & 15)) * kHW + vSlice;
      int kb = (lane & 16) ? 8 : 0;
#pragma unroll
      for (int r = 0; r < 8; ++r) {
        a0[r]     = p[(size_t)(kb + r) * posStride];
        a0[8 + r] = p[(size_t)(16 + kb + r) * posStride];
        a1[r]     = p[(size_t)(32 + kb + r) * posStride];
        a1[8 + r] = p[(size_t)(48 + kb + r) * posStride];
      }
#endif
    }
#pragma unroll
    for (int nt = 0; nt < 4; ++nt) {
      v8f o = {};
      o = wmma_bf16(a0, frag_b_packed(sAf + ((0 * 4 + nt) * 32) * 16), o);
      o = wmma_bf16(a1, frag_b_packed(sAf + ((1 * 4 + nt) * 32) * 16), o);
      int cBase = c0 + ((lane >> 4) << 3);
      int i = nt * 16 + (lane & 15);
#pragma unroll
      for (int d = 0; d < 8; ++d) {
        size_t idx = ((size_t)b * kC + cBase + d) * kHW +
                     (size_t)i * posStride + vSlice;
        if constexpr (DIR == 0) {
          outH[idx] = o[d];
        } else {
          out[idx] = g * (outH[idx] + o[d]) + x[idx];
        }
      }
    }
  }
}

// ---------------------------------------------------------------------------
extern "C" void kernel_launch(void* const* d_in, const int* in_sizes, int n_in,
                              void* d_out, int out_size, void* d_ws, size_t ws_size,
                              hipStream_t stream) {
  (void)in_sizes; (void)n_in; (void)out_size; (void)ws_size;
  const float* x     = (const float*)d_in[0];
  const float* Wq    = (const float*)d_in[1];
  const float* bq    = (const float*)d_in[2];
  const float* Wk    = (const float*)d_in[3];
  const float* bk    = (const float*)d_in[4];
  const float* Wv    = (const float*)d_in[5];
  const float* bv    = (const float*)d_in[6];
  const float* gamma = (const float*)d_in[7];
  float* out = (float*)d_out;
  char* ws = (char*)d_ws;

  const size_t qkBytes = (size_t)kB * kInter * kHW * sizeof(__bf16); // 8 MiB each
  const size_t vBytes  = (size_t)kB * kC * kHW * sizeof(__bf16);     // 64 MiB
  __bf16* qb = (__bf16*)ws;
  __bf16* kb = (__bf16*)(ws + qkBytes);
  __bf16* vb = (__bf16*)(ws + 2 * qkBytes);
  float*  oh = (float*)(ws + 2 * qkBytes + vBytes);                  // 128 MiB

  dim3 blk(256);
  qkv_gemm_kernel<<<dim3(kHW / 64, kInter / 32, kB), blk, 0, stream>>>(x, Wq, bq, qb, kInter);
  qkv_gemm_kernel<<<dim3(kHW / 64, kInter / 32, kB), blk, 0, stream>>>(x, Wk, bk, kb, kInter);
  qkv_gemm_kernel<<<dim3(kHW / 64, kC / 32, kB), blk, 0, stream>>>(x, Wv, bv, vb, kC);
  axis_attn_kernel<0><<<dim3(kW, kB), blk, 0, stream>>>(qb, kb, vb, oh, nullptr, nullptr, nullptr);
  axis_attn_kernel<1><<<dim3(kH, kB), blk, 0, stream>>>(qb, kb, vb, oh, x, gamma, out);
}
